// dialogue_VAE_60833916781092
// MI455X (gfx1250) — compile-verified, matrix operand order checked by reading, not run on hardware
//
#include <hip/hip_runtime.h>
#include <hip/hip_bf16.h>

// ---------------- problem constants ----------------
constexpr int BB = 32;          // batch
constexpr int TT = 512;         // time steps
constexpr int EE = 1024;        // embed
constexpr int HH = 1024;        // hidden
constexpr int H3 = 3 * HH;      // 3072 (gru gates)
constexpr int LL = 128;         // latent

// ---------------- WMMA fragment types ----------------
typedef __attribute__((ext_vector_type(16))) __bf16 v16bf;
typedef __attribute__((ext_vector_type(8)))  float  v8f;

union Frag16 { v16bf v; uint4 q[2]; };

// Load one 16x32 bf16 fragment (A-layout) from global memory.
// Lane supplies row (lane&15) and the K phase ((lane&16)?8:0); two 16-byte
// chunks at K offsets {0,16} relative to the phase.  B fragments use the
// identical pattern on W stored [N,K] row-major (i.e. B = W^T).
__device__ __forceinline__ v16bf load_frag(const __bf16* __restrict__ p) {
    Frag16 f;
    f.q[0] = *(const uint4*)(p);
    f.q[1] = *(const uint4*)(p + 16);
    return f.v;
}

// ---------------- fp32 -> bf16 conversion ----------------
__global__ void f2bf_kernel(const float* __restrict__ src, __bf16* __restrict__ dst, size_t n) {
    size_t i = (size_t)blockIdx.x * blockDim.x + threadIdx.x;
    size_t stride = (size_t)gridDim.x * blockDim.x;
    for (; i < n; i += stride) dst[i] = (__bf16)src[i];
}

// ---------------- stable rank (argsort of -lengths) ----------------
__global__ void sort_kernel(const int* __restrict__ lengths,
                            int* __restrict__ sorted_idx, int* __restrict__ rev_idx) {
    int b = threadIdx.x;
    if (b < BB) {
        int lb = lengths[b];
        int pos = 0;
        for (int j = 0; j < BB; ++j) {
            int lj = lengths[j];
            if (lj > lb || (lj == lb && j < b)) ++pos;
        }
        rev_idx[b] = pos;
        sorted_idx[pos] = b;
    }
}

// ---------------- generic bf16 WMMA GEMM:  D = act(A @ W^T + bias) --------
// A: [M,K] bf16 row-major.  W: [N,K] bf16 row-major.  D: [M,N] (bf16 or f32).
// Block = 256 threads = 8 waves.  Block tile = 128(M) x 64(N).
// Wave w owns M rows [blockIdx.y*128 + w*16, +16) and all 4 N sub-tiles.
//
// The 64x32 B tile (4 KB) is shared by all 8 waves, so it is staged into LDS
// once per block with gfx1250 async global->LDS loads (ASYNCcnt), double
// buffered; WMMA B fragments are then read with ds_load_b128.  Rows are
// padded to 80 B so the 16 lanes of a ds_load_b128 hit distinct banks.
constexpr int BROW  = 80;                // padded LDS row stride (bytes)
constexpr int BBUF  = 64 * BROW;         // one B tile buffer: 5120 B

template <bool RELU, bool OUT_BF16>
__global__ void gemm_bf16_kernel(const __bf16* __restrict__ A,
                                 const __bf16* __restrict__ W,
                                 const float*  __restrict__ bias,
                                 void* __restrict__ D,
                                 int N, int K) {
    __shared__ __align__(16) char smem[2 * BBUF];

    const int tid   = threadIdx.x;
    const int lane  = tid & 31;
    const int wave  = tid >> 5;
    const int mBase = blockIdx.y * 128 + wave * 16;
    const int nBase = blockIdx.x * 64;
    const int l15   = lane & 15;
    const int kph   = (lane & 16) ? 8 : 0;

    // loader mapping: thread i copies 16 bytes: B-tile row i>>2, K segment i&3
    const int lrow = tid >> 2;
    const int lseg = tid & 3;
    const __bf16* wsrc    = W + (size_t)(nBase + lrow) * K + lseg * 8;
    const unsigned lds_w  = (unsigned)(lrow * BROW + lseg * 16);

    v8f acc[4];
#pragma unroll
    for (int i = 0; i < 4; ++i) acc[i] = v8f{};

    const __bf16* aptr = A + (size_t)(mBase + l15) * K + kph;

    const int C = K / 32;     // K chunks of 32

    // prime buffer 0
    {
        unsigned laddr = lds_w;
        unsigned long long gaddr = (unsigned long long)(uintptr_t)wsrc;
        asm volatile("global_load_async_to_lds_b128 %0, %1, off"
                     :: "v"(laddr), "v"(gaddr) : "memory");
    }

    for (int c = 0; c < C; ++c) {
        if (c + 1 < C) {
            unsigned laddr = (unsigned)(((c + 1) & 1) * BBUF) + lds_w;
            unsigned long long gaddr =
                (unsigned long long)(uintptr_t)(wsrc + (size_t)(c + 1) * 32);
            asm volatile("global_load_async_to_lds_b128 %0, %1, off"
                         :: "v"(laddr), "v"(gaddr) : "memory");
            asm volatile("s_wait_asynccnt 0x1" ::: "memory");
        } else {
            asm volatile("s_wait_asynccnt 0x0" ::: "memory");
        }
        __syncthreads();    // chunk c is resident in LDS for all waves

        v16bf a = load_frag(aptr + c * 32);
        const char* bbase = smem + (c & 1) * BBUF + kph * 2;
#pragma unroll
        for (int nt = 0; nt < 4; ++nt) {
            const char* bp = bbase + (nt * 16 + l15) * BROW;
            Frag16 b;
            b.q[0] = *(const uint4*)(bp);
            b.q[1] = *(const uint4*)(bp + 32);
            acc[nt] = __builtin_amdgcn_wmma_f32_16x16x32_bf16(
                false, a, false, b.v, (short)0, acc[nt], false, false);
        }
        __syncthreads();    // all waves done with buf (c&1) before re-fill
    }

    const int rbase = mBase + kph;   // C/D row phase
#pragma unroll
    for (int nt = 0; nt < 4; ++nt) {
        int col = nBase + nt * 16 + l15;
        float bv = bias[col];
#pragma unroll
        for (int v = 0; v < 8; ++v) {
            float val = acc[nt][v] + bv;
            if (RELU) val = fmaxf(val, 0.0f);
            size_t idx = (size_t)(rbase + v) * N + col;
            if (OUT_BF16) ((__bf16*)D)[idx] = (__bf16)val;
            else          ((float*)D)[idx]  = val;
        }
    }
}

// ---------------- recurrent step: gh = h_bf16 @ Whh^T + bhh ----------------
// M = 32 (2 M-tiles), N = 3072, K = 1024.  Block = 256 thr = 8 waves.
// Wave w: M-tile = w&1, N strip = blockIdx.x*128 + (w>>1)*32 (2 sub-tiles).
// Latency-critical (serial chain of 512 steps): keep direct global loads,
// weights are L2-resident.
__global__ void gru_gh_kernel(const __bf16* __restrict__ hbf,
                              const __bf16* __restrict__ Whh,
                              const float*  __restrict__ bhh,
                              float* __restrict__ gh) {
    const int lane  = threadIdx.x & 31;
    const int wave  = threadIdx.x >> 5;
    const int mBase = (wave & 1) * 16;
    const int nBase = blockIdx.x * 128 + (wave >> 1) * 32;
    const int l15   = lane & 15;
    const int kph   = (lane & 16) ? 8 : 0;

    v8f acc[2];
    acc[0] = v8f{}; acc[1] = v8f{};

    const __bf16* aptr = hbf + (size_t)(mBase + l15) * HH + kph;
    const __bf16* wptr = Whh + (size_t)(nBase + l15) * HH + kph;

    for (int k = 0; k < HH; k += 32) {
        v16bf a = load_frag(aptr + k);
#pragma unroll
        for (int nt = 0; nt < 2; ++nt) {
            v16bf b = load_frag(wptr + (size_t)nt * 16 * HH + k);
            acc[nt] = __builtin_amdgcn_wmma_f32_16x16x32_bf16(
                false, a, false, b, (short)0, acc[nt], false, false);
        }
    }

    const int rbase = mBase + kph;
#pragma unroll
    for (int nt = 0; nt < 2; ++nt) {
        int col = nBase + nt * 16 + l15;
        float bv = bhh[col];
#pragma unroll
        for (int v = 0; v < 8; ++v)
            gh[(size_t)(rbase + v) * H3 + col] = acc[nt][v] + bv;
    }
}

// ---------------- GRU gate / state update -------------------------------
// gi: [B,T,3H] bf16 (precomputed x @ Wih^T + bih).  gh: [B,3H] f32.
// Updates h (f32 + bf16 shadow); decoder also writes y_t (bf16, 0 if padded).
template <bool WRITE_Y>
__global__ void gru_gate_kernel(const __bf16* __restrict__ gi,
                                const float*  __restrict__ gh,
                                float* __restrict__ h,
                                __bf16* __restrict__ hbf,
                                __bf16* __restrict__ y,
                                const int* __restrict__ lengths, int t) {
    int idx = blockIdx.x * blockDim.x + threadIdx.x;   // 0 .. B*H-1
    int b = idx >> 10;
    int j = idx & (HH - 1);

    const __bf16* g = gi + ((size_t)b * TT + t) * H3;
    float ir  = (float)g[j];
    float iz  = (float)g[HH + j];
    float inn = (float)g[2 * HH + j];

    const float* ghb = gh + (size_t)b * H3;
    float hr = ghb[j], hz = ghb[HH + j], hn = ghb[2 * HH + j];

    float hprev = h[idx];
    float r  = 1.0f / (1.0f + __expf(-(ir + hr)));
    float zg = 1.0f / (1.0f + __expf(-(iz + hz)));
    float n  = tanhf(inn + r * hn);
    float hnew = (1.0f - zg) * n + zg * hprev;

    bool valid = t < lengths[b];
    float hout = valid ? hnew : hprev;
    h[idx]   = hout;
    hbf[idx] = (__bf16)hout;
    if (WRITE_Y)
        y[((size_t)b * TT + t) * HH + j] = (__bf16)(valid ? hnew : 0.0f);
}

__global__ void zero_h_kernel(float* __restrict__ h, __bf16* __restrict__ hbf) {
    int idx = blockIdx.x * blockDim.x + threadIdx.x;
    h[idx] = 0.0f;
    hbf[idx] = (__bf16)0.0f;
}

// ---------------- VAE heads (sorted order outputs) ----------------------
// mean/logv/z are returned in SORTED order; gather h_enc through sorted_idx.
__global__ void heads_kernel(const float* __restrict__ hE,
                             const int* __restrict__ sorted_idx,
                             const float* __restrict__ Wm, const float* __restrict__ bm,
                             const float* __restrict__ Wv, const float* __restrict__ bv,
                             const float* __restrict__ eps,
                             float* __restrict__ mean_out,
                             float* __restrict__ logv_out,
                             float* __restrict__ z_out) {
    int i = blockIdx.x;      // sorted slot
    int l = threadIdx.x;     // 0..127
    int b = sorted_idx[i];
    const float* hv = hE + (size_t)b * HH;
    const float* wm = Wm + (size_t)l * HH;
    const float* wv = Wv + (size_t)l * HH;
    float sm = bm[l], sv = bv[l];
    for (int k = 0; k < HH; ++k) {
        float hk = hv[k];
        sm += hk * wm[k];
        sv += hk * wv[k];
    }
    mean_out[i * LL + l] = sm;
    logv_out[i * LL + l] = sv;
    z_out[i * LL + l] = eps[i * LL + l] * __expf(0.5f * sv) + sm;
}

// dec_h0 (unsorted order): hD[b] = z[rev_idx[b]] @ Wl^T + bl
__global__ void dech0_kernel(const float* __restrict__ z,
                             const int* __restrict__ rev_idx,
                             const float* __restrict__ Wl, const float* __restrict__ bl,
                             float* __restrict__ hD, __bf16* __restrict__ hDbf) {
    int b = blockIdx.x;
    const float* zr = z + (size_t)rev_idx[b] * LL;
    for (int j = threadIdx.x; j < HH; j += blockDim.x) {
        const float* wr = Wl + (size_t)j * LL;
        float s = bl[j];
        for (int l = 0; l < LL; ++l) s += zr[l] * wr[l];
        hD[(size_t)b * HH + j] = s;
        hDbf[(size_t)b * HH + j] = (__bf16)s;
    }
}

// ---------------- host side ------------------------------------------------
static inline size_t align_up(size_t x) { return (x + 255) & ~(size_t)255; }

extern "C" void kernel_launch(void* const* d_in, const int* in_sizes, int n_in,
                              void* d_out, int out_size, void* d_ws, size_t ws_size,
                              hipStream_t stream) {
    (void)in_sizes; (void)n_in; (void)out_size; (void)ws_size;

    const float* x    = (const float*)d_in[0];
    const float* eps  = (const float*)d_in[1];
    const float* W1   = (const float*)d_in[2];
    const float* b1   = (const float*)d_in[3];
    const float* W2   = (const float*)d_in[4];
    const float* b2   = (const float*)d_in[5];
    const float* WihE = (const float*)d_in[6];
    const float* WhhE = (const float*)d_in[7];
    const float* bihE = (const float*)d_in[8];
    const float* bhhE = (const float*)d_in[9];
    const float* WihD = (const float*)d_in[10];
    const float* WhhD = (const float*)d_in[11];
    const float* bihD = (const float*)d_in[12];
    const float* bhhD = (const float*)d_in[13];
    const float* Wm   = (const float*)d_in[14];
    const float* bm   = (const float*)d_in[15];
    const float* Wv   = (const float*)d_in[16];
    const float* bv   = (const float*)d_in[17];
    const float* Wl   = (const float*)d_in[18];
    const float* bl   = (const float*)d_in[19];
    const float* Wo   = (const float*)d_in[20];
    const float* bo   = (const float*)d_in[21];
    const int*   lengths = (const int*)d_in[22];

    float* out      = (float*)d_out;
    float* out_x    = out;
    float* out_y    = out_x + (size_t)BB * TT * EE;
    float* out_mean = out_y + (size_t)BB * TT * EE;
    float* out_logv = out_mean + BB * LL;
    float* out_z    = out_logv + BB * LL;

    // ---- workspace layout ----
    char* ws = (char*)d_ws;
    size_t off = 0;
    auto alloc = [&](size_t bytes) { char* p = ws + off; off = align_up(off + bytes); return p; };

    int*    sorted_idx = (int*)alloc(2 * BB * sizeof(int));
    int*    rev_idx    = sorted_idx + BB;
    __bf16* x_bf   = (__bf16*)alloc((size_t)BB * TT * EE * 2);   // reused later as ys_bf
    __bf16* h1_bf  = (__bf16*)alloc((size_t)BB * TT * HH * 2);
    __bf16* proj_bf= (__bf16*)alloc((size_t)BB * TT * HH * 2);
    __bf16* giE    = (__bf16*)alloc((size_t)BB * TT * H3 * 2);
    __bf16* giD    = (__bf16*)alloc((size_t)BB * TT * H3 * 2);
    __bf16* W1b    = (__bf16*)alloc((size_t)HH * EE * 2);
    __bf16* W2b    = (__bf16*)alloc((size_t)HH * HH * 2);
    __bf16* WihEb  = (__bf16*)alloc((size_t)H3 * HH * 2);
    __bf16* WhhEb  = (__bf16*)alloc((size_t)H3 * HH * 2);
    __bf16* WihDb  = (__bf16*)alloc((size_t)H3 * HH * 2);
    __bf16* WhhDb  = (__bf16*)alloc((size_t)H3 * HH * 2);
    __bf16* Wob    = (__bf16*)alloc((size_t)EE * HH * 2);
    float*  hE     = (float*)alloc((size_t)BB * HH * 4);
    __bf16* hEbf   = (__bf16*)alloc((size_t)BB * HH * 2);
    float*  hD     = (float*)alloc((size_t)BB * HH * 4);
    __bf16* hDbf   = (__bf16*)alloc((size_t)BB * HH * 2);
    float*  gh     = (float*)alloc((size_t)BB * H3 * 4);
    __bf16* ys_bf  = x_bf;   // x_bf is dead after proj1; reuse for decoder outputs

    const int M = BB * TT;   // 16384 rows for the big GEMMs

    // 0) pass-through output: x
    hipMemcpyAsync(out_x, x, (size_t)BB * TT * EE * sizeof(float),
                   hipMemcpyDeviceToDevice, stream);

    // 1) stable descending-length ranks
    sort_kernel<<<1, 32, 0, stream>>>(lengths, sorted_idx, rev_idx);

    // 2) fp32 -> bf16 conversions (activations + weights; weights stay L2-resident)
    f2bf_kernel<<<8192, 256, 0, stream>>>(x,    x_bf,  (size_t)BB * TT * EE);
    f2bf_kernel<<<2048, 256, 0, stream>>>(W1,   W1b,   (size_t)HH * EE);
    f2bf_kernel<<<2048, 256, 0, stream>>>(W2,   W2b,   (size_t)HH * HH);
    f2bf_kernel<<<4096, 256, 0, stream>>>(WihE, WihEb, (size_t)H3 * HH);
    f2bf_kernel<<<4096, 256, 0, stream>>>(WhhE, WhhEb, (size_t)H3 * HH);
    f2bf_kernel<<<4096, 256, 0, stream>>>(WihD, WihDb, (size_t)H3 * HH);
    f2bf_kernel<<<4096, 256, 0, stream>>>(WhhD, WhhDb, (size_t)H3 * HH);
    f2bf_kernel<<<2048, 256, 0, stream>>>(Wo,   Wob,   (size_t)EE * HH);

    // 3) shared projection: proj = (relu(x @ W1^T + b1)) @ W2^T + b2
    gemm_bf16_kernel<true,  true><<<dim3(HH / 64, M / 128), 256, 0, stream>>>(
        x_bf, W1b, b1, h1_bf, HH, EE);
    gemm_bf16_kernel<false, true><<<dim3(HH / 64, M / 128), 256, 0, stream>>>(
        h1_bf, W2b, b2, proj_bf, HH, HH);

    // 4) hoist the input-gate GEMMs out of the scan: gi = proj @ Wih^T + bih
    gemm_bf16_kernel<false, true><<<dim3(H3 / 64, M / 128), 256, 0, stream>>>(
        proj_bf, WihEb, bihE, giE, H3, HH);
    gemm_bf16_kernel<false, true><<<dim3(H3 / 64, M / 128), 256, 0, stream>>>(
        proj_bf, WihDb, bihD, giD, H3, HH);

    // 5) encoder GRU scan (h0 = 0); only the hidden-path GEMM is sequential
    zero_h_kernel<<<(BB * HH) / 256, 256, 0, stream>>>(hE, hEbf);
    for (int t = 0; t < TT; ++t) {
        gru_gh_kernel<<<H3 / 128, 256, 0, stream>>>(hEbf, WhhEb, bhhE, gh);
        gru_gate_kernel<false><<<(BB * HH) / 256, 256, 0, stream>>>(
            giE, gh, hE, hEbf, nullptr, lengths, t);
    }

    // 6) VAE heads (sorted-order outputs) + decoder h0 (scattered back)
    heads_kernel<<<BB, LL, 0, stream>>>(hE, sorted_idx, Wm, bm, Wv, bv, eps,
                                        out_mean, out_logv, out_z);
    dech0_kernel<<<BB, 256, 0, stream>>>(out_z, rev_idx, Wl, bl, hD, hDbf);

    // 7) decoder GRU scan; emits ys (bf16, zero-padded past each length)
    for (int t = 0; t < TT; ++t) {
        gru_gh_kernel<<<H3 / 128, 256, 0, stream>>>(hDbf, WhhDb, bhhD, gh);
        gru_gate_kernel<true><<<(BB * HH) / 256, 256, 0, stream>>>(
            giD, gh, hD, hDbf, ys_bf, lengths, t);
    }

    // 8) output projection straight into d_out (fp32)
    gemm_bf16_kernel<false, false><<<dim3(EE / 64, M / 128), 256, 0, stream>>>(
        ys_bf, Wob, bo, out_y, EE, HH);
}
